// MrT5LayerSelfAttention_79912161509624
// MI455X (gfx1250) — matrix operand order repaired and checked by reading, not compiled, not run on hardware
//
#include <hip/hip_runtime.h>
#include <hip/hip_bf16.h>

// ---- problem constants (match reference) ----
#define B_   2
#define S_   2048
#define D_   1024
#define H_   16
#define DK_  64
#define HDK_ 1024   // H_*DK_
#define NBIAS 4096  // bias table width per head (index = key - query + 2048)

typedef _Float16 v16h __attribute__((ext_vector_type(16)));
typedef _Float16 v8h  __attribute__((ext_vector_type(8)));
typedef float    v8f  __attribute__((ext_vector_type(8)));
typedef int      v4i  __attribute__((ext_vector_type(4)));

static __device__ __forceinline__ v8f wmma_f16(v16h a, v16h b, v8f c) {
    return __builtin_amdgcn_wmma_f32_16x16x32_f16(false, a, false, b, (short)0, c, false, false);
}

// ---- gfx1250 async global->LDS (ASYNCcnt) with portable fallback ----
#if defined(__has_builtin)
#if __has_builtin(__builtin_amdgcn_global_load_async_to_lds_b128)
#define HAS_ASYNC_LDS 1
#endif
#endif

static __device__ __forceinline__ void async_copy16(_Float16* lds_dst, const _Float16* g_src) {
#ifdef HAS_ASYNC_LDS
    __builtin_amdgcn_global_load_async_to_lds_b128(
        (__attribute__((address_space(1))) v4i*)g_src,
        (__attribute__((address_space(3))) v4i*)lds_dst,
        /*imm offset*/0, /*cpol*/0);
#else
    *(v8h*)lds_dst = *(const v8h*)g_src;
#endif
}
static __device__ __forceinline__ void async_wait0() {
#ifdef HAS_ASYNC_LDS
#if __has_builtin(__builtin_amdgcn_s_wait_asynccnt)
    __builtin_amdgcn_s_wait_asynccnt(0);
#else
    asm volatile("s_wait_asynccnt 0" ::: "memory");
#endif
#endif
}

// ------------------------------------------------------------------
// 1) RMSNorm (T5 LayerNorm): x16 = f16( h * rsqrt(mean(h^2)+eps) * w )
// ------------------------------------------------------------------
__global__ __launch_bounds__(256) void k_rmsnorm(const float* __restrict__ hs,
                                                 const float* __restrict__ ln_w,
                                                 _Float16* __restrict__ x16) {
    const int row = blockIdx.x;                 // 0..B*S-1
    const float* r = hs + (size_t)row * D_;
    float vals[4];
    float ssq = 0.f;
#pragma unroll
    for (int i = 0; i < 4; ++i) {
        vals[i] = r[threadIdx.x + i * 256];
        ssq += vals[i] * vals[i];
    }
#pragma unroll
    for (int m = 1; m < 32; m <<= 1) ssq += __shfl_xor(ssq, m, 32);
    __shared__ float red[8];
    if ((threadIdx.x & 31) == 0) red[threadIdx.x >> 5] = ssq;
    __syncthreads();
    float tot = 0.f;
#pragma unroll
    for (int i = 0; i < 8; ++i) tot += red[i];
    const float scale = rsqrtf(tot * (1.0f / D_) + 1e-6f);
#pragma unroll
    for (int i = 0; i < 4; ++i) {
        const int c = threadIdx.x + i * 256;
        x16[(size_t)row * D_ + c] = (_Float16)(vals[i] * scale * ln_w[c]);
    }
}

// ------------------------------------------------------------------
// 2) Weights f32 -> f16, transposed: wT[n][k] = w[k][n]
// ------------------------------------------------------------------
__global__ __launch_bounds__(256) void k_convert_w(const float* __restrict__ qw,
                                                   const float* __restrict__ kw,
                                                   const float* __restrict__ vw,
                                                   const float* __restrict__ ow,
                                                   _Float16* __restrict__ wq,
                                                   _Float16* __restrict__ wk,
                                                   _Float16* __restrict__ wv,
                                                   _Float16* __restrict__ wo) {
    const size_t idx = (size_t)blockIdx.x * 256 + threadIdx.x;  // over 1024*1024
    const int k = (int)(idx >> 10);
    const int n = (int)(idx & 1023);
    const size_t dst = (size_t)n * 1024 + k;
    wq[dst] = (_Float16)qw[idx];
    wk[dst] = (_Float16)kw[idx];
    wv[dst] = (_Float16)vw[idx];
    wo[dst] = (_Float16)ow[idx];
}

// ------------------------------------------------------------------
// 3) Relative-position bias table: btab[h][p] = rel_emb[bucket(p-2048)][h]
// ------------------------------------------------------------------
__global__ __launch_bounds__(256) void k_bias_table(const float* __restrict__ rel_emb,
                                                    float* __restrict__ btab) {
    const int idx = blockIdx.x * 256 + threadIdx.x;  // H_ * NBIAS = 65536
    const int h = idx >> 12;
    const int p = idx & (NBIAS - 1);
    const int rel = p - 2048;                        // key - query
    int bucket = (rel > 0) ? 16 : 0;                 // num_buckets/2 = 16
    int n = rel < 0 ? -rel : rel;
    int bval;
    if (n < 8) {                                     // max_exact = 8
        bval = n;
    } else {
        float l = __logf((float)n * 0.125f) * (1.0f / 2.7725887222397811f) * 8.0f; // /log(16)*8
        bval = 8 + (int)l;
        if (bval > 15) bval = 15;
    }
    btab[idx] = rel_emb[(bucket + bval) * H_ + h];
}

// ------------------------------------------------------------------
// 4) QKV projection GEMM with async-LDS-staged B tiles (double buffered).
//    grid: (N/64, M/128, 3); each wave: 16x64 tile, K-loop step 32.
// ------------------------------------------------------------------
__global__ __launch_bounds__(256) void k_qkv_gemm(const _Float16* __restrict__ x16,
                                                  const _Float16* __restrict__ wqT,
                                                  const _Float16* __restrict__ wkT,
                                                  const _Float16* __restrict__ wvT,
                                                  _Float16* __restrict__ q16,
                                                  _Float16* __restrict__ k16,
                                                  _Float16* __restrict__ vT16) {
    __shared__ __align__(16) _Float16 bbuf[2][64 * 32];   // [n][k] tile, 4KB x2
    const int tid  = threadIdx.x;
    const int wave = tid >> 5;
    const int lane = tid & 31;
    const int lhalf = lane >> 4, lmod = lane & 15;
    const int which = blockIdx.z;
    const _Float16* wT = (which == 0) ? wqT : (which == 1) ? wkT : wvT;
    const int m0 = blockIdx.y * 128 + wave * 16;
    const int n0 = blockIdx.x * 64;

    // staging slice for this thread: row bn of B tile, 8 halfs at bs
    const int bn = tid >> 2;
    const int bs = (tid & 3) * 8;
    const _Float16* gB = wT + (size_t)(n0 + bn) * D_ + bs;

    const _Float16* arow = x16 + (size_t)(m0 + lmod) * D_;
    v8f acc[4] = {};

    async_copy16(&bbuf[0][bn * 32 + bs], gB);             // prologue: chunk 0

    for (int k0 = 0, it = 0; k0 < D_; k0 += 32, ++it) {
        const int cur = it & 1;
        async_wait0();
        __syncthreads();                                   // bbuf[cur] ready for all waves
        if (k0 + 32 < D_)
            async_copy16(&bbuf[cur ^ 1][bn * 32 + bs], gB + k0 + 32);

        // A fragment (16x32 f16): lane holds row lmod, K = lhalf*16 + 0..15
        union { v16h v; v8h h2[2]; } ua;
        ua.h2[0] = *(const v8h*)(arow + k0 + lhalf * 16);
        ua.h2[1] = *(const v8h*)(arow + k0 + 8 + lhalf * 16);
        const v16h a = ua.v;
#pragma unroll
        for (int j = 0; j < 4; ++j) {
            // B fragment from LDS: col = j*16+lmod, K = lhalf*16 + 0..15
            const v16h b = *(const v16h*)&bbuf[cur][(j * 16 + lmod) * 32 + lhalf * 16];
            acc[j] = wmma_f16(a, b, acc[j]);
        }
    }
    // store: C layout -> row m = m0 + v + 8*lhalf, col n = n0 + j*16 + lmod
#pragma unroll
    for (int j = 0; j < 4; ++j) {
#pragma unroll
        for (int v = 0; v < 8; ++v) {
            const int m = m0 + v + 8 * lhalf;
            const int n = n0 + j * 16 + lmod;
            const int b = m >> 11, s = m & (S_ - 1);
            const int h = n >> 6, dk = n & (DK_ - 1);
            const _Float16 val = (_Float16)acc[j][v];
            if (which == 0)
                q16[((size_t)(b * H_ + h) * S_ + s) * DK_ + dk] = val;
            else if (which == 1)
                k16[((size_t)(b * H_ + h) * S_ + s) * DK_ + dk] = val;
            else
                vT16[((size_t)(b * H_ + h) * DK_ + dk) * S_ + s] = val;  // transposed
        }
    }
}

// ------------------------------------------------------------------
// 5) Flash attention with softmax1. K/V chunks staged once per block
//    via async global->LDS (shared by 8 waves), double buffered.
//    grid: (S/128, B*H), 256 threads (8 waves, 16 queries each)
// ------------------------------------------------------------------
__global__ __launch_bounds__(256) void k_attention(const _Float16* __restrict__ q16,
                                                   const _Float16* __restrict__ k16,
                                                   const _Float16* __restrict__ vT16,
                                                   const float* __restrict__ btab,
                                                   const float* __restrict__ gate,
                                                   _Float16* __restrict__ ao16) {
    __shared__ __align__(16) _Float16 kbuf[2][32 * 64];  // [key][dk]   4KB x2
    __shared__ __align__(16) _Float16 vbuf[2][64 * 32];  // [dk][key]   4KB x2
    __shared__ __align__(16) _Float16 pbuf[8][16 * 32];  // per-wave probs
    const int tid  = threadIdx.x;
    const int wave = tid >> 5;
    const int lane = tid & 31;
    const int lhalf = lane >> 4, lmod = lane & 15;
    const int bh = blockIdx.y;
    const int b = bh >> 4, h = bh & 15;
    const int qbase = (blockIdx.x * 8 + wave) * 16;

    const _Float16* Qb = q16 + (size_t)(b * H_ + h) * S_ * DK_;
    const _Float16* Kb = k16 + (size_t)(b * H_ + h) * S_ * DK_;
    const _Float16* Vb = vT16 + (size_t)(b * H_ + h) * DK_ * S_;
    const float* bt = btab + h * NBIAS;
    const float* gb = gate + b * S_;
    _Float16* myP = &pbuf[wave][0];

    // staging slices: K chunk = 32 keys x 64 dk; V chunk = 64 dk x 32 keys
    const int kkey = tid >> 3, kseg = (tid & 7) * 8;     // K: row kkey, 8 halfs at kseg
    const int vd   = tid >> 2, vseg = (tid & 3) * 8;     // V: row vd,   8 halfs at vseg

    // Q fragments for dk chunks 0..31, 32..63 (kept in registers)
    v16h qa[2];
    {
        const _Float16* qr = Qb + (size_t)(qbase + lmod) * DK_;
#pragma unroll
        for (int c = 0; c < 2; ++c) {
            union { v16h v; v8h h2[2]; } u;
            u.h2[0] = *(const v8h*)(qr + c * 32 + lhalf * 16);
            u.h2[1] = *(const v8h*)(qr + c * 32 + 8 + lhalf * 16);
            qa[c] = u.v;
        }
    }

    float mrow[8], lrow[8];
    v8f oacc[4] = {};
#pragma unroll
    for (int v = 0; v < 8; ++v) { mrow[v] = -1e30f; lrow[v] = 0.f; }

    // prologue: stage chunk 0
    async_copy16(&kbuf[0][kkey * 64 + kseg], Kb + (size_t)kkey * DK_ + kseg);
    async_copy16(&vbuf[0][vd * 32 + vseg],  Vb + (size_t)vd * S_ + vseg);

    for (int kc = 0, it = 0; kc < S_; kc += 32, ++it) {
        const int cur = it & 1;
        async_wait0();
        __syncthreads();                                  // chunk `cur` visible to all waves
        if (kc + 32 < S_) {
            async_copy16(&kbuf[cur ^ 1][kkey * 64 + kseg],
                         Kb + (size_t)(kc + 32 + kkey) * DK_ + kseg);
            async_copy16(&vbuf[cur ^ 1][vd * 32 + vseg],
                         Vb + (size_t)vd * S_ + (kc + 32) + vseg);
        }

        // ---- scores: 16 queries x 32 keys = two 16x16 WMMA tiles ----
        v8f sc[2];
#pragma unroll
        for (int t = 0; t < 2; ++t) {
            const _Float16* kr = &kbuf[cur][(t * 16 + lmod) * 64];
            const v16h kb0 = *(const v16h*)(kr + lhalf * 16);        // dk 0..31
            const v16h kb1 = *(const v16h*)(kr + 32 + lhalf * 16);   // dk 32..63
            v8f s = {};
            s = wmma_f16(qa[0], kb0, s);
            s = wmma_f16(qa[1], kb1, s);
            sc[t] = s;
        }
        // ---- add relative bias + delete gate ----
#pragma unroll
        for (int t = 0; t < 2; ++t) {
            const int key = kc + t * 16 + lmod;
            const float g = gb[key];
#pragma unroll
            for (int v = 0; v < 8; ++v) {
                const int qrow = qbase + v + 8 * lhalf;
                sc[t][v] += bt[key - qrow + 2048] + g;
            }
        }
        // ---- online softmax1 update (row = v + 8*lhalf) ----
        float mnew[8], alpha[8];
#pragma unroll
        for (int v = 0; v < 8; ++v) {
            float mx = fmaxf(sc[0][v], sc[1][v]);
#pragma unroll
            for (int m = 1; m < 16; m <<= 1) mx = fmaxf(mx, __shfl_xor(mx, m, 32));
            mnew[v] = fmaxf(mrow[v], mx);
            alpha[v] = __expf(mrow[v] - mnew[v]);
            mrow[v] = mnew[v];
        }
#pragma unroll
        for (int t = 0; t < 2; ++t)
#pragma unroll
            for (int v = 0; v < 8; ++v)
                sc[t][v] = __expf(sc[t][v] - mnew[v]);
#pragma unroll
        for (int v = 0; v < 8; ++v) {
            float sm = sc[0][v] + sc[1][v];
#pragma unroll
            for (int m = 1; m < 16; m <<= 1) sm += __shfl_xor(sm, m, 32);
            lrow[v] = lrow[v] * alpha[v] + sm;
        }
#pragma unroll
        for (int j = 0; j < 4; ++j)
#pragma unroll
            for (int v = 0; v < 8; ++v) oacc[j][v] *= alpha[v];

        // ---- probs: C layout -> per-wave LDS tile -> A fragment ----
        // (same-wave DS ops are in-order; no cross-wave sharing => no barrier)
#pragma unroll
        for (int t = 0; t < 2; ++t)
#pragma unroll
            for (int v = 0; v < 8; ++v)
                myP[(v + 8 * lhalf) * 32 + t * 16 + lmod] = (_Float16)sc[t][v];
        union { v16h v; v8h h2[2]; } pu;
        pu.h2[0] = *(const v8h*)&myP[lmod * 32 + lhalf * 16];
        pu.h2[1] = *(const v8h*)&myP[lmod * 32 + 8 + lhalf * 16];
        const v16h pa = pu.v;
        // ---- O += P @ V : 4 WMMAs over 64 dk columns (V from LDS) ----
#pragma unroll
        for (int j = 0; j < 4; ++j) {
            const v16h vb = *(const v16h*)&vbuf[cur][(j * 16 + lmod) * 32 + lhalf * 16];
            oacc[j] = wmma_f16(pa, vb, oacc[j]);
        }
    }

    // softmax1 denominator: l + exp(-m); store attn_out f16 as [b][s][h*64+dk]
    float inv[8];
#pragma unroll
    for (int v = 0; v < 8; ++v) inv[v] = 1.0f / (lrow[v] + __expf(-mrow[v]));
#pragma unroll
    for (int j = 0; j < 4; ++j)
#pragma unroll
        for (int v = 0; v < 8; ++v) {
            const int qrow = qbase + v + 8 * lhalf;
            const int n = h * DK_ + j * 16 + lmod;
            ao16[(size_t)(b * S_ + qrow) * HDK_ + n] = (_Float16)(oacc[j][v] * inv[v]);
        }
}

// ------------------------------------------------------------------
// 6) Output projection + residual (async-LDS-staged B tiles)
// ------------------------------------------------------------------
__global__ __launch_bounds__(256) void k_out_proj(const _Float16* __restrict__ ao16,
                                                  const _Float16* __restrict__ woT,
                                                  const float* __restrict__ hs,
                                                  float* __restrict__ out) {
    __shared__ __align__(16) _Float16 bbuf[2][64 * 32];
    const int tid  = threadIdx.x;
    const int wave = tid >> 5;
    const int lane = tid & 31;
    const int lhalf = lane >> 4, lmod = lane & 15;
    const int m0 = blockIdx.y * 128 + wave * 16;
    const int n0 = blockIdx.x * 64;

    const int bn = tid >> 2;
    const int bs = (tid & 3) * 8;
    const _Float16* gB = woT + (size_t)(n0 + bn) * HDK_ + bs;

    const _Float16* arow = ao16 + (size_t)(m0 + lmod) * HDK_;
    v8f acc[4] = {};

    async_copy16(&bbuf[0][bn * 32 + bs], gB);

    for (int k0 = 0, it = 0; k0 < HDK_; k0 += 32, ++it) {
        const int cur = it & 1;
        async_wait0();
        __syncthreads();
        if (k0 + 32 < HDK_)
            async_copy16(&bbuf[cur ^ 1][bn * 32 + bs], gB + k0 + 32);

        union { v16h v; v8h h2[2]; } ua;
        ua.h2[0] = *(const v8h*)(arow + k0 + lhalf * 16);
        ua.h2[1] = *(const v8h*)(arow + k0 + 8 + lhalf * 16);
        const v16h a = ua.v;
#pragma unroll
        for (int j = 0; j < 4; ++j) {
            const v16h b = *(const v16h*)&bbuf[cur][(j * 16 + lmod) * 32 + lhalf * 16];
            acc[j] = wmma_f16(a, b, acc[j]);
        }
    }
#pragma unroll
    for (int j = 0; j < 4; ++j)
#pragma unroll
        for (int v = 0; v < 8; ++v) {
            const int m = m0 + v + 8 * lhalf;
            const int n = n0 + j * 16 + lmod;
            out[(size_t)m * D_ + n] = hs[(size_t)m * D_ + n] + acc[j][v];
        }
}

// ------------------------------------------------------------------
extern "C" void kernel_launch(void* const* d_in, const int* in_sizes, int n_in,
                              void* d_out, int out_size, void* d_ws, size_t ws_size,
                              hipStream_t stream) {
    const float* hs      = (const float*)d_in[0];  // [B,S,D]
    const float* gate    = (const float*)d_in[1];  // [B,S,1]
    const float* ln_w    = (const float*)d_in[2];  // [D]
    const float* q_w     = (const float*)d_in[3];  // [D, H*DK]
    const float* k_w     = (const float*)d_in[4];
    const float* v_w     = (const float*)d_in[5];
    const float* o_w     = (const float*)d_in[6];  // [H*DK, D]
    const float* rel_emb = (const float*)d_in[7];  // [NUM_BUCKETS, H]
    float* out = (float*)d_out;

    char* ws = (char*)d_ws;
    const size_t MB = 1u << 20;
    _Float16* x16  = (_Float16*)(ws + 0);          //  8 MB  (B*S*D f16)
    _Float16* wq16 = (_Float16*)(ws + 8 * MB);     //  2 MB  transposed
    _Float16* wk16 = (_Float16*)(ws + 10 * MB);    //  2 MB
    _Float16* wv16 = (_Float16*)(ws + 12 * MB);    //  2 MB
    _Float16* wo16 = (_Float16*)(ws + 14 * MB);    //  2 MB
    _Float16* q16  = (_Float16*)(ws + 16 * MB);    //  8 MB  [b,h,s,dk]
    _Float16* k16  = (_Float16*)(ws + 24 * MB);    //  8 MB  [b,h,s,dk]
    _Float16* vT16 = (_Float16*)(ws + 32 * MB);    //  8 MB  [b,h,dk,s]
    _Float16* ao16 = (_Float16*)(ws + 40 * MB);    //  8 MB  [b,s,h*dk]
    float*    btab = (float*)   (ws + 48 * MB);    // 256 KB [H,4096]

    k_rmsnorm   <<<B_ * S_, 256, 0, stream>>>(hs, ln_w, x16);
    k_convert_w <<<(D_ * HDK_) / 256, 256, 0, stream>>>(q_w, k_w, v_w, o_w,
                                                        wq16, wk16, wv16, wo16);
    k_bias_table<<<(H_ * NBIAS) / 256, 256, 0, stream>>>(rel_emb, btab);
    k_qkv_gemm  <<<dim3(HDK_ / 64, (B_ * S_) / 128, 3), 256, 0, stream>>>(
        x16, wq16, wk16, wv16, q16, k16, vT16);
    k_attention <<<dim3(S_ / 128, B_ * H_), 256, 0, stream>>>(
        q16, k16, vT16, btab, gate, ao16);
    k_out_proj  <<<dim3(D_ / 64, (B_ * S_) / 128), 256, 0, stream>>>(
        ao16, wo16, hs, out);
}
